// MultiLevelDynamicGraphEncoder_49443663511669
// MI455X (gfx1250) — matrix operand order repaired and checked by reading, not compiled, not run on hardware
//
#include <hip/hip_runtime.h>
#include <hip/hip_bf16.h>

typedef _Float16 v16h __attribute__((ext_vector_type(16)));
typedef float    v8f  __attribute__((ext_vector_type(8)));

#define N_NODES 2048
#define T_STEPS 20
#define O_DIM   64
#define K_NBR   32

union F16Frag { v16h v; uint4 q[2]; };
union H2U     { _Float16 h[2]; unsigned u; };

// Native-rate activations: v_exp_f32 + v_rcp_f32, no OCML range-check branches.
__device__ __forceinline__ float fast_sigmoid(float x) {
    return __builtin_amdgcn_rcpf(1.0f + __expf(-x));
}
__device__ __forceinline__ float fast_tanh(float x) {
    // tanh(x) = 2*sigmoid(2x) - 1
    return fmaf(2.0f, __builtin_amdgcn_rcpf(1.0f + __expf(-2.0f * x)), -1.0f);
}

__device__ __forceinline__ v8f wmma_acc(v16h a, v16h b, v8f c) {
    return __builtin_amdgcn_wmma_f32_16x16x32_f16(false, a, false, b, (short)0, c,
                                                  false, false);
}

// ---------------------------------------------------------------------------
// Phase 1: graph conv + max-pool over K neighbors.
// One wave per (t,n): lane = neighbor k. Each lane owns output channels
// o = 2*lane, 2*lane+1; neighbor diffs are broadcast via uniform-index shfl
// (v_readlane). Result stored as f16 (WMMA A operand precision).
// pooled layout: [T][N][64] f16.
// ---------------------------------------------------------------------------
__global__ void __launch_bounds__(256) conv_pool_kernel(
    const float* __restrict__ X,      // [N][T][2]
    const int*   __restrict__ As,     // [3][T][N][K]
    const float* __restrict__ Wc,     // [64][2]
    const float* __restrict__ bc,     // [64]
    _Float16*    __restrict__ pooled, // [T][N][64]
    int level)
{
    const int wid  = threadIdx.x >> 5;
    const int lane = threadIdx.x & 31;
    const int pair = blockIdx.x * 8 + wid;           // 0 .. T*N-1
    const int t = pair / N_NODES;
    const int n = pair % N_NODES;

    const long aoff = (((long)level * T_STEPS + t) * N_NODES + n) * K_NBR + lane;
    const int idx = As[aoff];

    const float xc0 = X[(long)n * (T_STEPS * 2) + t * 2 + 0];
    const float xc1 = X[(long)n * (T_STEPS * 2) + t * 2 + 1];
    const float d0  = X[(long)idx * (T_STEPS * 2) + t * 2 + 0] - xc0;
    const float d1  = X[(long)idx * (T_STEPS * 2) + t * 2 + 1] - xc1;

    const int o0 = 2 * lane, o1 = 2 * lane + 1;
    const float w00 = Wc[o0 * 2], w01 = Wc[o0 * 2 + 1];
    const float w10 = Wc[o1 * 2], w11 = Wc[o1 * 2 + 1];

    float m0 = -3.4e38f, m1 = -3.4e38f;
#pragma unroll
    for (int k = 0; k < K_NBR; ++k) {
        float b0 = __shfl(d0, k, 32);
        float b1 = __shfl(d1, k, 32);
        m0 = fmaxf(m0, fmaf(b0, w00, b1 * w01));
        m1 = fmaxf(m1, fmaf(b0, w10, b1 * w11));
    }
    m0 += bc[o0];
    m1 += bc[o1];

    H2U p;
    p.h[0] = (_Float16)m0;
    p.h[1] = (_Float16)m1;
    ((unsigned*)pooled)[(long)pair * 32 + lane] = p.u;   // f16 pair at o=2*lane
}

// ---------------------------------------------------------------------------
// One LSTM timestep: gates = Ax@Wih^T (+ Ah@Whh^T if USE_H) + b, then the
// elementwise cell update. Fully unrolled, no runtime branches.
// ---------------------------------------------------------------------------
template <bool USE_H, bool FIRST>
__device__ __forceinline__ void lstm_cell_step(
    const F16Frag* __restrict__ ax,
    const F16Frag* __restrict__ ah,
    const unsigned (*__restrict__ wf)[16][2][32][8],
    int lane, int lanelo, int half, int nodeBase, int t,
    const float* __restrict__ bR,
    v8f* __restrict__ hC, v8f* __restrict__ cC,
    float* __restrict__ Xout)
{
#pragma unroll
    for (int jj = 0; jj < 4; ++jj) {
        v8f acc[4];
#pragma unroll
        for (int g = 0; g < 4; ++g) {
            const int tile = g * 4 + jj;     // i: tiles 0-3, f: 4-7, g: 8-11, o: 12-15
            v8f a = {0.f, 0.f, 0.f, 0.f, 0.f, 0.f, 0.f, 0.f};
#pragma unroll
            for (int cc = 0; cc < 2; ++cc) {
                F16Frag bi;
                bi.q[0] = *(const uint4*)(&wf[0][tile][cc][lane][0]);
                bi.q[1] = *(const uint4*)(&wf[0][tile][cc][lane][4]);
                a = wmma_acc(ax[cc].v, bi.v, a);
                if (USE_H) {
                    F16Frag bh;
                    bh.q[0] = *(const uint4*)(&wf[1][tile][cc][lane][0]);
                    bh.q[1] = *(const uint4*)(&wf[1][tile][cc][lane][4]);
                    a = wmma_acc(ah[cc].v, bh.v, a);
                }
            }
            acc[g] = a;
        }
        // ---- elementwise LSTM cell update: C/D layouts align across gates ----
        const int colb = jj * 16 + lanelo;
#pragma unroll
        for (int e = 0; e < 8; ++e) {
            const float ig = fast_sigmoid(acc[0][e] + bR[0 * 4 + jj]);
            const float fg = fast_sigmoid(acc[1][e] + bR[1 * 4 + jj]);
            const float gg = fast_tanh   (acc[2][e] + bR[2 * 4 + jj]);
            const float og = fast_sigmoid(acc[3][e] + bR[3 * 4 + jj]);
            const float cv = fg * cC[jj][e] + ig * gg;
            const float hv = og * fast_tanh(cv);
            cC[jj][e] = cv;
            hC[jj][e] = hv;

            const int n = nodeBase + e + 8 * half;
            const long xoi = ((long)n * T_STEPS + t) * 64 + colb;
            if (FIRST) Xout[xoi] = hv;
            else       Xout[xoi] = fmaxf(Xout[xoi], hv);
        }
    }
}

__device__ __forceinline__ void load_ax(F16Frag* ax, const _Float16* __restrict__ pooled,
                                        int t, int rowg, int half)
{
    const _Float16* prow = pooled + ((long)t * N_NODES + rowg) * 64;
#pragma unroll
    for (int cc = 0; cc < 2; ++cc) {
        ax[cc].q[0] = *(const uint4*)(prow + cc * 32 + half * 8);        // VGPR 0-3
        ax[cc].q[1] = *(const uint4*)(prow + cc * 32 + 16 + half * 8);   // VGPR 4-7
    }
}

// ---------------------------------------------------------------------------
// Phase 2: LSTM over T=20 steps with v_wmma_f32_16x16x32_f16.
// 128 threads = 4 waves; each wave owns 16 nodes, keeps h/c in registers
// (C/D-matrix layout: 4 col-tiles x v8f). t=0 is peeled (h==0 -> Wih only),
// so the steady-state loop is a branch-free stream of 64 WMMAs per step.
// The per-step h relayout uses a PER-WAVE LDS region: same-wave LDS ops are
// in-order (DScnt), so no workgroup barrier is needed in the hot loop.
// ---------------------------------------------------------------------------
template <bool FIRST>
__global__ void __launch_bounds__(128) dgcn_lstm_kernel(
    const _Float16* __restrict__ pooled, // [T][N][64] f16
    const float* __restrict__ Wih,       // [256][64]
    const float* __restrict__ Whh,       // [256][64]
    const float* __restrict__ bih,       // [256]
    const float* __restrict__ bhh,       // [256]
    float* __restrict__ Xout,            // [N][T][64]
    float* __restrict__ Hout,            // [N][64]
    float* __restrict__ Cout)            // [N][64]
{
    // [matrix][col-tile][k-chunk][lane][vgpr] : f16x2 fragments, 64 KB
    __shared__ __align__(16) unsigned  wfrag[2][16][2][32][8];
    __shared__ __align__(16) _Float16  htmp[4][16][64];     // 8 KB, per-wave regions

    const int tid = threadIdx.x;

    // ---- one-time: swizzle both weight matrices into B-fragment layout ----
    for (int lin = tid; lin < 2 * 16 * 2 * 32 * 8; lin += 128) {
        const int v  =  lin        & 7;
        const int ln = (lin >> 3)  & 31;
        const int cc = (lin >> 8)  & 1;
        const int tj = (lin >> 9)  & 15;
        const int m  =  lin >> 13;
        const int col = tj * 16 + (ln & 15);               // gate column (N of B)
        const int kk  = cc * 32 + (ln >> 4) * 8 + (v >> 2) * 8 + 2 * v;  // K index
        const float* W = m ? Whh : Wih;                    // B[k][col] = W[col][k]
        H2U p;
        p.h[0] = (_Float16)W[col * 64 + kk];
        p.h[1] = (_Float16)W[col * 64 + kk + 1];
        ((unsigned*)wfrag)[lin] = p.u;
    }
    __syncthreads();   // cross-wave: weights must be visible to all waves

    const int wid    = tid >> 5;
    const int lane   = tid & 31;
    const int lanelo = lane & 15;
    const int half   = lane >> 4;
    const int nodeBase = blockIdx.x * 64 + wid * 16;
    const int rowg     = nodeBase + lanelo;

    // per-lane fused bias b = bih + bhh for the 16 (gate, col-tile) combos
    float bR[16];
#pragma unroll
    for (int g = 0; g < 4; ++g)
#pragma unroll
        for (int jj = 0; jj < 4; ++jj) {
            const int c = g * 64 + jj * 16 + lanelo;
            bR[g * 4 + jj] = bih[c] + bhh[c];
        }

    const v8f zz = {0.f, 0.f, 0.f, 0.f, 0.f, 0.f, 0.f, 0.f};
    v8f hC[4], cC[4];
#pragma unroll
    for (int jj = 0; jj < 4; ++jj) { hC[jj] = zz; cC[jj] = zz; }

    F16Frag ax[2], ah[2];

    // ---- t = 0 peeled: h == 0, only the Wih matmul contributes ----
    load_ax(ax, pooled, 0, rowg, half);
    lstm_cell_step<false, FIRST>(ax, ah, wfrag, lane, lanelo, half, nodeBase, 0,
                                 bR, hC, cC, Xout);

    // ---- steady state: branch-free 64-WMMA stream per step ----
    for (int t = 1; t < T_STEPS; ++t) {
        load_ax(ax, pooled, t, rowg, half);

        // re-layout h (C/D layout) into A-fragments via per-wave LDS bounce.
        // Same-wave LDS ops are in-order; region is wave-private -> no barrier.
#pragma unroll
        for (int jj = 0; jj < 4; ++jj)
#pragma unroll
            for (int e = 0; e < 8; ++e)
                htmp[wid][e + 8 * half][jj * 16 + lanelo] = (_Float16)hC[jj][e];
#pragma unroll
        for (int cc = 0; cc < 2; ++cc) {
            ah[cc].q[0] = *(const uint4*)(&htmp[wid][lanelo][cc * 32 + half * 8]);
            ah[cc].q[1] = *(const uint4*)(&htmp[wid][lanelo][cc * 32 + 16 + half * 8]);
        }

        lstm_cell_step<true, FIRST>(ax, ah, wfrag, lane, lanelo, half, nodeBase, t,
                                    bR, hC, cC, Xout);
    }

    // ---- final h/c, max across levels ----
#pragma unroll
    for (int jj = 0; jj < 4; ++jj)
#pragma unroll
        for (int e = 0; e < 8; ++e) {
            const int n = nodeBase + e + 8 * half;
            const long oi = (long)n * 64 + jj * 16 + lanelo;
            if (FIRST) {
                Hout[oi] = hC[jj][e];
                Cout[oi] = cC[jj][e];
            } else {
                Hout[oi] = fmaxf(Hout[oi], hC[jj][e]);
                Cout[oi] = fmaxf(Cout[oi], cC[jj][e]);
            }
        }
}

// ---------------------------------------------------------------------------
extern "C" void kernel_launch(void* const* d_in, const int* in_sizes, int n_in,
                              void* d_out, int out_size, void* d_ws, size_t ws_size,
                              hipStream_t stream)
{
    const float* X   = (const float*)d_in[0];
    const int*   As  = (const int*)  d_in[1];
    const float* Wc  = (const float*)d_in[2];
    const float* bc  = (const float*)d_in[3];
    const float* Wih = (const float*)d_in[4];
    const float* Whh = (const float*)d_in[5];
    const float* bih = (const float*)d_in[6];
    const float* bhh = (const float*)d_in[7];

    float* out  = (float*)d_out;
    float* Xout = out;                                         // N*T*64
    float* Hout = out + (long)N_NODES * T_STEPS * O_DIM;       // N*64
    float* Cout = Hout + (long)N_NODES * O_DIM;                // N*64

    _Float16* pooled = (_Float16*)d_ws;                        // T*N*64 f16 = 5.24 MB

    const int convBlocks = (T_STEPS * N_NODES) / 8;            // 8 waves per 256-thr block
    const int lstmBlocks = N_NODES / 64;                       // 64 nodes per 4-wave block

    for (int l = 0; l < 3; ++l) {
        conv_pool_kernel<<<convBlocks, 256, 0, stream>>>(X, As, Wc, bc, pooled, l);
        if (l == 0)
            dgcn_lstm_kernel<true><<<lstmBlocks, 128, 0, stream>>>(
                pooled, Wih, Whh, bih, bhh, Xout, Hout, Cout);
        else
            dgcn_lstm_kernel<false><<<lstmBlocks, 128, 0, stream>>>(
                pooled, Wih, Whh, bih, bhh, Xout, Hout, Cout);
    }
}
